// CareGptOssAttentionHF_46059229283038
// MI455X (gfx1250) — compile-verified, matrix operand order checked by reading, not run on hardware
//
#include <hip/hip_runtime.h>
#include <hip/hip_bf16.h>

#define S_LEN   2048
#define HIDN    2048
#define NHEAD   16
#define DQK     128
#define DV      64
#define RLAT    512
#define KVA     576      // RLAT + ROPE
#define ROPE_D  64
#define SWIN    512
#define SCALEF  0.08838834764831845f   // 1/sqrt(128)
#define NEGINF  (-1e30f)

#define MBLK 128
#define NBLK 64
#define KCH  32

typedef __bf16 bf16_t;
typedef __attribute__((ext_vector_type(16))) __bf16 v16bf;
typedef __attribute__((ext_vector_type(8)))  __bf16 v8bf;
typedef __attribute__((ext_vector_type(8)))  float  v8f;
typedef __attribute__((ext_vector_type(4)))  unsigned int u32x4;
typedef __attribute__((ext_vector_type(8)))  int i32x8;
typedef __attribute__((ext_vector_type(4)))  int i32x4;

static __device__ __forceinline__ bf16_t f2bf(float f) {
  union { float f; unsigned u; } v; v.f = f;
  unsigned r = v.u + 0x7FFFu + ((v.u >> 16) & 1u);
  union { unsigned short s; bf16_t b; } o; o.s = (unsigned short)(r >> 16);
  return o.b;
}

// A/B fragment loader for 16x16x32 bf16 WMMA, row-major source (global or LDS).
// Caller passes p = rowBase + kOff where kOff = (lane>>4)*8; chunks at +0 and +16.
static __device__ __forceinline__ v16bf load_frag(const bf16_t* __restrict__ p) {
  v8bf lo = *(const v8bf*)(p);
  v8bf hi = *(const v8bf*)(p + 16);
  v16bf r;
#pragma unroll
  for (int i = 0; i < 8; ++i) { r[i] = lo[i]; r[8 + i] = hi[i]; }
  return r;
}

// ---------------------------------------------------------- TDM 2-D tile load
// D# per CDNA5 ISA.
// Loads a (tile_h x tile_w) bf16 tile (row stride = row_stride elements) from
// gptr into LDS offset lds_off, row-major compact (tile_w*2 bytes per row).
static __device__ __forceinline__ void tdm_load_2d(unsigned lds_off,
                                                   const bf16_t* gptr,
                                                   unsigned tile_w, unsigned tile_h,
                                                   unsigned row_stride) {
  unsigned long long ga = (unsigned long long)gptr;
  unsigned tensor_w = row_stride;      // >= tile_w, tiles never clipped
  unsigned tensor_h = 0x00FFFFFFu;     // huge: no OOB clipping in dim1
  u32x4 g0;
  g0[0] = 1u;                                            // count=1, user mode
  g0[1] = lds_off;                                       // lds_addr
  g0[2] = (unsigned)(ga & 0xFFFFFFFFu);                  // global_addr[31:0]
  g0[3] = (unsigned)((ga >> 32) & 0x01FFFFFFu)           // global_addr[56:32]
        | (2u << 30);                                    // type=2 ("image")
  i32x8 g1;
  g1[0] = (int)(1u << 16);                               // data_size=1 -> 2 bytes
  g1[1] = (int)((tensor_w & 0xFFFFu) << 16);             // tensor_dim0[15:0]
  g1[2] = (int)((tensor_w >> 16) | ((tensor_h & 0xFFFFu) << 16));
  g1[3] = (int)((tensor_h >> 16) | (tile_w << 16));      // tile_dim0
  g1[4] = (int)(tile_h);                                 // tile_dim1 (tile_dim2=0)
  g1[5] = (int)(row_stride);                             // tensor_dim0_stride[31:0]
  g1[6] = 0;                                             // stride0 hi / stride1 lo
  g1[7] = 0;
  i32x4 z4 = {0, 0, 0, 0};
  i32x8 z8 = {0, 0, 0, 0, 0, 0, 0, 0};
  __builtin_amdgcn_tensor_load_to_lds(g0, g1, z4, z4, z8, 0);
}

// ---------------------------------------------------------------- conversions
__global__ void cvt_f32_bf16(const float* __restrict__ src, bf16_t* __restrict__ dst, long n) {
  long i = (long)blockIdx.x * blockDim.x + threadIdx.x;
  long stride = (long)gridDim.x * blockDim.x;
  for (; i < n; i += stride) dst[i] = f2bf(src[i]);
}

// w_vc (H, R, DV) -> w_vcT (H, DV, R)
__global__ void wvc_transpose(const float* __restrict__ wvc, bf16_t* __restrict__ out) {
  long i = (long)blockIdx.x * blockDim.x + threadIdx.x;
  long n = (long)NHEAD * DV * RLAT;
  long stride = (long)gridDim.x * blockDim.x;
  for (; i < n; i += stride) {
    int r = (int)(i % RLAT);
    int d = (int)((i / RLAT) % DV);
    int h = (int)(i / ((long)RLAT * DV));
    out[i] = f2bf(wvc[((long)h * RLAT + r) * DV + d]);
  }
}

// -------------------------------------------------- TDM + LDS tiled GEMM
// C[M,N] = A[M,K] @ W[N,K]^T + bias (f32 out).
// Block = 8 waves, 128x64 output tile; TDM double-buffers 128x32 A and 64x32 W
// tiles into LDS (wave 0 drives the DMA, all waves consume via ds_load_b128).
__global__ __launch_bounds__(256) void gemm_bf16_tdm(const bf16_t* __restrict__ A,
                                                     const bf16_t* __restrict__ W,
                                                     const float* __restrict__ bias,
                                                     float* __restrict__ Cout,
                                                     int M, int N, int K) {
  __shared__ bf16_t ldsA[2][MBLK * KCH];
  __shared__ bf16_t ldsB[2][NBLK * KCH];
  int lane = threadIdx.x & 31;
  int wave = threadIdx.x >> 5;
  int nB = blockIdx.x * NBLK;
  int mB = blockIdx.y * MBLK;
  int r = lane & 15, kOff = (lane >> 4) * 8;
  int nk = K / KCH;

  if (wave == 0) {
    tdm_load_2d((unsigned)(size_t)&ldsA[0][0], A + (long)mB * K, KCH, MBLK, K);
    tdm_load_2d((unsigned)(size_t)&ldsB[0][0], W + (long)nB * K, KCH, NBLK, K);
    __builtin_amdgcn_s_wait_tensorcnt(0);
  }
  __syncthreads();

  v8f acc[4] = {};
  for (int kc = 0; kc < nk; ++kc) {
    int buf = kc & 1;
    if (wave == 0 && (kc + 1) < nk) {
      // previous reads of buf^1 finished before the barrier that ended iter kc-1
      tdm_load_2d((unsigned)(size_t)&ldsA[buf ^ 1][0],
                  A + (long)mB * K + (kc + 1) * KCH, KCH, MBLK, K);
      tdm_load_2d((unsigned)(size_t)&ldsB[buf ^ 1][0],
                  W + (long)nB * K + (kc + 1) * KCH, KCH, NBLK, K);
    }
    v16bf a = load_frag(&ldsA[buf][(wave * 16 + r) * KCH + kOff]);
#pragma unroll
    for (int t = 0; t < 4; ++t) {
      v16bf b = load_frag(&ldsB[buf][(t * 16 + r) * KCH + kOff]);
      acc[t] = __builtin_amdgcn_wmma_f32_16x16x32_bf16(false, a, false, b, (short)0,
                                                       acc[t], false, false);
    }
    if (wave == 0 && (kc + 1) < nk) __builtin_amdgcn_s_wait_tensorcnt(0);
    __syncthreads();
  }

  int mBase = mB + wave * 16 + (lane >> 4) * 8;
#pragma unroll
  for (int t = 0; t < 4; ++t) {
    int n = nB + t * 16 + r;
    float bv = bias ? bias[n] : 0.0f;
#pragma unroll
    for (int j = 0; j < 8; ++j)
      Cout[(long)(mBase + j) * N + n] = acc[t][j] + bv;
  }
}

// --------------------------------------------------- latent -> k_nope / v
// z=0: k_nope[h] = lat @ w_kc[h]^T -> kf[h][s][0:64]
// z=1: v[h]      = lat @ w_vc[h]   -> vT[h][d][s]     (via pre-transposed w_vcT)
__global__ void knope_v_kernel(const bf16_t* __restrict__ lat, const bf16_t* __restrict__ wkc,
                               const bf16_t* __restrict__ wvcT, bf16_t* __restrict__ kf,
                               bf16_t* __restrict__ vT) {
  int lane = threadIdx.x;
  int mT = blockIdx.x, h = blockIdx.y, mode = blockIdx.z;
  int r = lane & 15, kOff = (lane >> 4) * 8;
  const bf16_t* W = (mode == 0 ? wkc : wvcT) + (long)h * DV * RLAT;
  const bf16_t* aRow = lat + (long)(mT * 16 + r) * RLAT + kOff;
  v8f acc[4] = {};
  for (int k0 = 0; k0 < RLAT; k0 += 32) {
    v16bf a = load_frag(aRow + k0);
#pragma unroll
    for (int t = 0; t < 4; ++t) {
      v16bf b = load_frag(W + (long)(t * 16 + r) * RLAT + k0 + kOff);
      acc[t] = __builtin_amdgcn_wmma_f32_16x16x32_bf16(false, a, false, b, (short)0,
                                                       acc[t], false, false);
    }
  }
  int mBase = mT * 16 + (lane >> 4) * 8;
#pragma unroll
  for (int t = 0; t < 4; ++t) {
    int n = t * 16 + r;
#pragma unroll
    for (int j = 0; j < 8; ++j) {
      int m = mBase + j;
      bf16_t val = f2bf(acc[t][j]);
      if (mode == 0) kf[((long)h * S_LEN + m) * DQK + n] = val;
      else           vT[((long)h * DV + n) * S_LEN + m]  = val;
    }
  }
}

// ------------------------------------------------------- RoPE + pack + cast
__global__ void rope_pack(const float* __restrict__ qlin, const float* __restrict__ latent,
                          const float* __restrict__ cosb, const float* __restrict__ sinb,
                          bf16_t* __restrict__ qf, bf16_t* __restrict__ kf,
                          bf16_t* __restrict__ latbf) {
  int s = blockIdx.x, t = threadIdx.x;
  latbf[(long)s * RLAT + t] = f2bf(latent[(long)s * KVA + t]);
  if (t < ROPE_D) {
    float c = cosb[(long)s * ROPE_D + t], sn = sinb[(long)s * ROPE_D + t];
    float kr = latent[(long)s * KVA + RLAT + t];
    float krot = (t < 32) ? -latent[(long)s * KVA + RLAT + t + 32]
                          :  latent[(long)s * KVA + RLAT + t - 32];
    bf16_t kvb = f2bf(kr * c + krot * sn);
#pragma unroll
    for (int h = 0; h < NHEAD; ++h) {
      kf[((long)h * S_LEN + s) * DQK + 64 + t] = kvb;
      const float* qrow = qlin + (long)s * HIDN + h * DQK;
      qf[((long)h * S_LEN + s) * DQK + t] = f2bf(qrow[t]);
      float qr = qrow[64 + t];
      float qrot = (t < 32) ? -qrow[64 + t + 32] : qrow[64 + t - 32];
      qf[((long)h * S_LEN + s) * DQK + 64 + t] = f2bf(qr * c + qrot * sn);
    }
  }
}

// ------------------------------------------------------------ attention
// One wave per (head, 16-query tile); 32-key blocks; sink folded into init state.
__global__ void attn_kernel(const bf16_t* __restrict__ qf, const bf16_t* __restrict__ kf,
                            const bf16_t* __restrict__ vT, const float* __restrict__ sinks,
                            bf16_t* __restrict__ attn) {
  __shared__ bf16_t p_lds[16 * 32];
  int lane = threadIdx.x;
  int mT = blockIdx.x, h = blockIdx.y;
  int m0 = mT * 16;
  int r = lane & 15, half = lane >> 4, kOff = half * 8;

  v16bf qa[4];
  const bf16_t* qrow = qf + ((long)h * S_LEN + m0 + r) * DQK + kOff;
#pragma unroll
  for (int i = 0; i < 4; ++i) qa[i] = load_frag(qrow + i * 32);

  float sink = sinks[h];
  float m_run[8], l_run[8];
#pragma unroll
  for (int j = 0; j < 8; ++j) { m_run[j] = sink; l_run[j] = 1.0f; }
  v8f acc[4] = {};

  int kLo = m0 - (SWIN - 1); if (kLo < 0) kLo = 0; kLo &= ~31;
  int kHi = m0 + 16;

  for (int kb = kLo; kb < kHi; kb += 32) {
    v8f sc[2] = {};
#pragma unroll
    for (int nb = 0; nb < 2; ++nb) {
      const bf16_t* krow = kf + ((long)h * S_LEN + kb + nb * 16 + r) * DQK + kOff;
#pragma unroll
      for (int i = 0; i < 4; ++i) {
        v16bf kfr = load_frag(krow + i * 32);
        sc[nb] = __builtin_amdgcn_wmma_f32_16x16x32_bf16(false, qa[i], false, kfr, (short)0,
                                                         sc[nb], false, false);
      }
    }
    float tmax[8];
#pragma unroll
    for (int j = 0; j < 8; ++j) tmax[j] = NEGINF;
#pragma unroll
    for (int nb = 0; nb < 2; ++nb)
#pragma unroll
      for (int j = 0; j < 8; ++j) {
        int qi = m0 + j + half * 8;
        int kj = kb + nb * 16 + r;
        float sv = sc[nb][j] * SCALEF;
        bool ok = (kj <= qi) && (qi - kj < SWIN);
        sv = ok ? sv : NEGINF;
        sc[nb][j] = sv;
        tmax[j] = fmaxf(tmax[j], sv);
      }
#pragma unroll
    for (int j = 0; j < 8; ++j) {
      tmax[j] = fmaxf(tmax[j], __shfl_xor(tmax[j], 1, 32));
      tmax[j] = fmaxf(tmax[j], __shfl_xor(tmax[j], 2, 32));
      tmax[j] = fmaxf(tmax[j], __shfl_xor(tmax[j], 4, 32));
      tmax[j] = fmaxf(tmax[j], __shfl_xor(tmax[j], 8, 32));
    }
    float alpha[8], rsum[8];
#pragma unroll
    for (int j = 0; j < 8; ++j) {
      float nm = fmaxf(m_run[j], tmax[j]);
      alpha[j] = __expf(m_run[j] - nm);
      m_run[j] = nm;
    }
#pragma unroll
    for (int nb = 0; nb < 2; ++nb)
#pragma unroll
      for (int j = 0; j < 8; ++j) sc[nb][j] = __expf(sc[nb][j] - m_run[j]);
#pragma unroll
    for (int j = 0; j < 8; ++j) {
      float v = sc[0][j] + sc[1][j];
      v += __shfl_xor(v, 1, 32);
      v += __shfl_xor(v, 2, 32);
      v += __shfl_xor(v, 4, 32);
      v += __shfl_xor(v, 8, 32);
      rsum[j] = v;
    }
#pragma unroll
    for (int j = 0; j < 8; ++j) l_run[j] = l_run[j] * alpha[j] + rsum[j];
#pragma unroll
    for (int t = 0; t < 4; ++t)
#pragma unroll
      for (int j = 0; j < 8; ++j) acc[t][j] *= alpha[j];

    // C-layout P tile -> LDS (row-major 16x32 bf16) -> A-layout fragment
#pragma unroll
    for (int nb = 0; nb < 2; ++nb)
#pragma unroll
      for (int j = 0; j < 8; ++j)
        p_lds[(j + half * 8) * 32 + nb * 16 + r] = f2bf(sc[nb][j]);
    __syncthreads();
    v16bf pf = load_frag(p_lds + r * 32 + kOff);
#pragma unroll
    for (int t = 0; t < 4; ++t) {
      const bf16_t* vrow = vT + ((long)h * DV + t * 16 + r) * S_LEN + kb + kOff;
      v16bf vf = load_frag(vrow);
      acc[t] = __builtin_amdgcn_wmma_f32_16x16x32_bf16(false, pf, false, vf, (short)0,
                                                       acc[t], false, false);
    }
    __syncthreads();
  }
#pragma unroll
  for (int t = 0; t < 4; ++t)
#pragma unroll
    for (int j = 0; j < 8; ++j) {
      int s = m0 + j + half * 8;
      int col = h * DV + t * 16 + r;
      attn[(long)s * (NHEAD * DV) + col] = f2bf(acc[t][j] / l_run[j]);
    }
}

// ------------------------------------------------------------------ launch
extern "C" void kernel_launch(void* const* d_in, const int* in_sizes, int n_in,
                              void* d_out, int out_size, void* d_ws, size_t ws_size,
                              hipStream_t stream) {
  const float* hidden = (const float*)d_in[0];
  const float* cosb   = (const float*)d_in[1];
  const float* sinb   = (const float*)d_in[2];
  const float* Wq     = (const float*)d_in[3];
  const float* bq     = (const float*)d_in[4];
  const float* Wo     = (const float*)d_in[5];
  const float* bo     = (const float*)d_in[6];
  const float* Wkva   = (const float*)d_in[7];
  const float* bkva   = (const float*)d_in[8];
  const float* wkc    = (const float*)d_in[9];
  const float* wvc    = (const float*)d_in[10];
  const float* sinks  = (const float*)d_in[11];

  char* ws = (char*)d_ws;
  size_t off = 0;
  auto take = [&](size_t bytes) { size_t o = off; off += (bytes + 255) & ~(size_t)255; return o; };

  bf16_t* hid_bf  = (bf16_t*)(ws + take((size_t)S_LEN * HIDN * 2));
  bf16_t* wq_bf   = (bf16_t*)(ws + take((size_t)HIDN * HIDN * 2));
  bf16_t* wkva_bf = (bf16_t*)(ws + take((size_t)KVA * HIDN * 2));
  bf16_t* wkc_bf  = (bf16_t*)(ws + take((size_t)NHEAD * DV * RLAT * 2));
  bf16_t* wvcT_bf = (bf16_t*)(ws + take((size_t)NHEAD * DV * RLAT * 2));
  bf16_t* wo_bf   = (bf16_t*)(ws + take((size_t)HIDN * NHEAD * DV * 2));
  float*  q_f32   = (float*) (ws + take((size_t)S_LEN * HIDN * 4));
  float*  lat_f32 = (float*) (ws + take((size_t)S_LEN * KVA * 4));
  bf16_t* lat_bf  = (bf16_t*)(ws + take((size_t)S_LEN * RLAT * 2));
  bf16_t* qf      = (bf16_t*)(ws + take((size_t)NHEAD * S_LEN * DQK * 2));
  bf16_t* kf      = (bf16_t*)(ws + take((size_t)NHEAD * S_LEN * DQK * 2));
  bf16_t* vT      = (bf16_t*)(ws + take((size_t)NHEAD * DV * S_LEN * 2));
  bf16_t* attn_bf = (bf16_t*)(ws + take((size_t)S_LEN * NHEAD * DV * 2));

  // 1) fp32 -> bf16 conversions
  cvt_f32_bf16<<<4096, 256, 0, stream>>>(hidden, hid_bf, (long)S_LEN * HIDN);
  cvt_f32_bf16<<<4096, 256, 0, stream>>>(Wq, wq_bf, (long)HIDN * HIDN);
  cvt_f32_bf16<<<2048, 256, 0, stream>>>(Wkva, wkva_bf, (long)KVA * HIDN);
  cvt_f32_bf16<<<1024, 256, 0, stream>>>(wkc, wkc_bf, (long)NHEAD * DV * RLAT);
  cvt_f32_bf16<<<2048, 256, 0, stream>>>(Wo, wo_bf, (long)HIDN * NHEAD * DV);
  wvc_transpose<<<1024, 256, 0, stream>>>(wvc, wvcT_bf);

  // 2) q = hidden @ Wq^T + bq    (2048 x 2048 x 2048), TDM-staged
  {
    dim3 g(HIDN / NBLK, S_LEN / MBLK);
    gemm_bf16_tdm<<<g, 256, 0, stream>>>(hid_bf, wq_bf, bq, q_f32, S_LEN, HIDN, HIDN);
  }
  // 3) latent = hidden @ Wkva^T + bkva   (2048 x 576 x 2048), TDM-staged
  {
    dim3 g(KVA / NBLK, S_LEN / MBLK);
    gemm_bf16_tdm<<<g, 256, 0, stream>>>(hid_bf, wkva_bf, bkva, lat_f32, S_LEN, KVA, HIDN);
  }

  // 4) RoPE + pack qf/kf(rope half) + latent->bf16
  rope_pack<<<S_LEN, RLAT, 0, stream>>>(q_f32, lat_f32, cosb, sinb, qf, kf, lat_bf);

  // 5) k_nope (-> kf[...,0:64]) and v (-> vT), per head
  {
    dim3 g(S_LEN / 16, NHEAD, 2);
    knope_v_kernel<<<g, 32, 0, stream>>>(lat_bf, wkc_bf, wvcT_bf, kf, vT);
  }

  // 6) sliding-window flash attention with sink
  {
    dim3 g(S_LEN / 16, NHEAD);
    attn_kernel<<<g, 32, 0, stream>>>(qf, kf, vT, sinks, attn_bf);
  }

  // 7) final = attn @ Wo^T + bo   (2048 x 2048 x 1024), f32 output, TDM-staged
  {
    dim3 g(HIDN / NBLK, S_LEN / MBLK);
    gemm_bf16_tdm<<<g, 256, 0, stream>>>(attn_bf, wo_bf, bo, (float*)d_out,
                                         S_LEN, HIDN, NHEAD * DV);
  }
}
